// HyperMoE_29686813950417
// MI455X (gfx1250) — matrix operand order repaired and checked by reading, not compiled, not run on hardware
//
#include <hip/hip_runtime.h>
#include <hip/hip_bf16.h>

// ---------------------------------------------------------------------------
// HyperMoE: D/U hypernet weight-gen + per-batch GEMMs on CDNA5 WMMA (bf16).
//   x        : [8, 2048, 1024] f32
//   k        : [8, 128]        f32
//   D_matrix : [1024*128, 128] f32   -> D[b] = (D_matrix @ k[b]) as [1024,128]
//   U_matrix : [128*1024, 128] f32   -> U[b] = (U_matrix @ k[b]) as [128,1024]
//   h = relu(x @ D);  out = h @ U
// bf16 WMMA w/ f32 accumulation; 512-thread blocks (16 wave32s) on a 128x128
// tile; double-buffered LDS, 1 barrier/iter. A-tile: f32 global -> native bf16
// convert -> LDS. B-tile: GLOBAL_LOAD_ASYNC_TO_LDS_B128 DMA (ASYNCcnt), no
// VGPR round-trip. sched_group_barrier pins DSwrite->VMEM->DSread->WMMA burst.
// ---------------------------------------------------------------------------

typedef __attribute__((ext_vector_type(16))) __bf16 v16bf;
typedef __attribute__((ext_vector_type(8)))  float  v8f;
typedef __attribute__((ext_vector_type(4)))  float  f32x4;
typedef __attribute__((ext_vector_type(4)))  __bf16 bf16x4;

#define BATCH   8
#define SEQ     2048
#define IN_DIM  1024
#define LOW     128
#define OUT_DIM 1024

#define BM 128
#define BN 128
#define BK 32
#define LDS_STRIDE 40   // ushorts per row: 80 B (16B-aligned rows, bank-spread)

__device__ __forceinline__ unsigned short f2bf(float f) {
    return __builtin_bit_cast(unsigned short, (__bf16)f);   // native cvt, RNE
}

__device__ __forceinline__ void wait_async0() {
#if __has_builtin(__builtin_amdgcn_s_wait_asynccnt)
    __builtin_amdgcn_s_wait_asynccnt(0);
#else
    asm volatile("s_wait_asynccnt 0x0" ::: "memory");
#endif
}

// ---------------------------------------------------------------------------
// Kernel 1: per-batch weight generation, written TRANSPOSED in bf16.
//   Dt[b][l][d] = sum_j D_matrix[d*128+l][j] * k[b][j]    ([8][128][1024])
//   Ut[b][o][l] = sum_j U_matrix[l*1024+o][j] * k[b][j]   ([8][1024][128])
// ---------------------------------------------------------------------------
__global__ __launch_bounds__(256)
void gen_weights_kernel(const float* __restrict__ Dm, const float* __restrict__ Um,
                        const float* __restrict__ kv,
                        unsigned short* __restrict__ Dt, unsigned short* __restrict__ Ut)
{
    __shared__ float ks[BATCH * LOW];          // 4 KB
    const int t = threadIdx.x;
    for (int i = t; i < BATCH * LOW; i += 256) ks[i] = kv[i];
    __syncthreads();

    const long gid   = (long)blockIdx.x * 256 + t;       // 0 .. 262143
    const bool isU   = gid >= (long)IN_DIM * LOW;
    const long p     = isU ? gid - (long)IN_DIM * LOW : gid;
    const f32x4* r4  = reinterpret_cast<const f32x4*>((isU ? Um : Dm) + p * LOW);

    float acc[BATCH];
#pragma unroll
    for (int b = 0; b < BATCH; ++b) acc[b] = 0.0f;

    for (int j = 0; j < LOW / 4; ++j) {
        const f32x4 a = __builtin_nontemporal_load(r4 + j);  // streamed once
#pragma unroll
        for (int b = 0; b < BATCH; ++b) {
            const f32x4 kk = *reinterpret_cast<const f32x4*>(&ks[b * LOW + j * 4]);
            acc[b] = fmaf(a.x, kk.x, fmaf(a.y, kk.y, fmaf(a.z, kk.z, fmaf(a.w, kk.w, acc[b]))));
        }
    }

    if (!isU) {
        const int d = (int)(p >> 7), l = (int)(p & 127);
#pragma unroll
        for (int b = 0; b < BATCH; ++b)
            Dt[((long)b * LOW + l) * IN_DIM + d] = f2bf(acc[b]);
    } else {
        const int l = (int)(p >> 10), o = (int)(p & 1023);
#pragma unroll
        for (int b = 0; b < BATCH; ++b)
            Ut[((long)b * OUT_DIM + o) * LOW + l] = f2bf(acc[b]);
    }
}

// ---------------------------------------------------------------------------
// Kernel 2/3: batched GEMM  C[b] = op(A[b] @ Bt[b]^T),  A f32 [S x K],
// Bt bf16 [N x K] (pre-transposed), C f32 [S x N].
// 512 threads = 16 wave32s; 128x128 tile; wave (rg,cg) owns rows rg*16..+15
// and column tiles cg*4..cg*4+3. BK=32; double-buffered LDS; async-DMA B.
// ---------------------------------------------------------------------------
template<bool RELU>
__global__ __launch_bounds__(512)
void wmma_gemm_kernel(const float* __restrict__ A, const unsigned short* __restrict__ Bt,
                      float* __restrict__ C, const int K, const int Ntot)
{
    __shared__ alignas(16) unsigned short As[2][BM * LDS_STRIDE];   // 2 x 10 KB
    __shared__ alignas(16) unsigned short Bs[2][BN * LDS_STRIDE];   // 2 x 10 KB

    const int t    = threadIdx.x;
    const int lane = t & 31;
    const int wv   = t >> 5;          // wave id 0..15
    const int rg   = wv & 7;          // row group (16 rows)
    const int cg   = wv >> 3;         // column group (4 WMMA tiles)
    const int half = lane >> 4;       // lane half (ISA fragment layouts)
    const int l16  = lane & 15;
    const int b    = blockIdx.z;
    const long m0  = (long)blockIdx.x * BM;
    const long n0  = (long)blockIdx.y * BN;

    const float*          Ab = A  + ((long)b * SEQ  + m0) * K;
    const unsigned short* Bb = Bt + ((long)b * Ntot + n0) * K;
    float*                Cb = C  + ((long)b * SEQ  + m0) * Ntot + n0;

    v8f acc[4];
#pragma unroll
    for (int i = 0; i < 4; ++i) acc[i] = (v8f){0.f,0.f,0.f,0.f,0.f,0.f,0.f,0.f};

    // staging: 4 threads per row, 8 elements (one 16B chunk) each
    const int sr = t >> 2, sq = t & 3;
    const f32x4* Ag = reinterpret_cast<const f32x4*>(Ab + (long)sr * K + sq * 8);
    const unsigned long long Bb64  = (unsigned long long)Bb;
    const unsigned           Bboff = (unsigned)(((long)sr * K + sq * 8) * 2); // bytes
    const int mrow  = rg * 16 + l16;
    const int iters = K / BK;

    f32x4 ga[2];      // in-flight A chunk (f32, converted at LDS-store time)

    auto loadG = [&](int i) {
        const f32x4* ap = Ag + (long)i * (BK / 4);
        if (RELU) {   // x is read exactly once -> non-temporal
            ga[0] = __builtin_nontemporal_load(ap + 0);
            ga[1] = __builtin_nontemporal_load(ap + 1);
        } else {      // h is L2-resident
            ga[0] = ap[0];
            ga[1] = ap[1];
        }
    };

    auto storeLDS_A = [&](int p) {      // native f32->bf16 vector converts (RNE)
        const uint2 lo = __builtin_bit_cast(uint2, __builtin_convertvector(ga[0], bf16x4));
        const uint2 hi = __builtin_bit_cast(uint2, __builtin_convertvector(ga[1], bf16x4));
        *reinterpret_cast<uint4*>(&As[p][sr * LDS_STRIDE + sq * 8]) =
            make_uint4(lo.x, lo.y, hi.x, hi.y);
    };

    // B tile: per-lane 16B DMA straight into LDS (ASYNCcnt-tracked, no VGPRs)
    auto asyncB = [&](int i, int p) {
        const unsigned lds  = (unsigned)(unsigned long long)(void*)
                              &Bs[p][sr * LDS_STRIDE + sq * 8];
        const unsigned goff = Bboff + (unsigned)(i * BK * 2);
        asm volatile("global_load_async_to_lds_b128 %0, %1, %2"
                     :: "v"(lds), "v"(goff), "s"(Bb64)
                     : "memory");
    };

    union Frag { v16bf v; uint4 q[2]; };

    // ---- pipeline prologue -------------------------------------------------
    asyncB(0, 0);
    loadG(0);
    storeLDS_A(0);
    if (iters > 1) loadG(1);
    wait_async0();
    __syncthreads();

    // ---- main loop: 1 barrier per K-step -----------------------------------
    for (int i = 0; i < iters; ++i) {
        const int p = i & 1;

        // stage tile i+1 into other buffer (A via regs, B via async DMA),
        // then prefetch A tile i+2 -- all overlap the WMMA section below.
        if (i + 1 < iters) {
            storeLDS_A(p ^ 1);
            asyncB(i + 1, p ^ 1);
        }
        if (i + 2 < iters) loadG(i + 2);

        // A fragment (ISA 16-bit A 16x32): VGPR0-3 <- K=8h..8h+7, VGPR4-7 <- 16+8h..
        Frag fa;
        fa.q[0] = *reinterpret_cast<const uint4*>(&As[p][mrow * LDS_STRIDE + half * 8]);
        fa.q[1] = *reinterpret_cast<const uint4*>(&As[p][mrow * LDS_STRIDE + 16 + half * 8]);

        // 4 B fragments in distinct registers -> one dscnt wait, WMMA burst
        Frag fb[4];
#pragma unroll
        for (int j = 0; j < 4; ++j) {
            const unsigned short* bp =
                &Bs[p][((cg * 4 + j) * 16 + l16) * LDS_STRIDE + half * 8];
            fb[j].q[0] = reinterpret_cast<const uint4*>(bp)[0];
            fb[j].q[1] = reinterpret_cast<const uint4*>(bp + 16)[0];
        }
#pragma unroll
        for (int j = 0; j < 4; ++j) {
            acc[j] = __builtin_amdgcn_wmma_f32_16x16x32_bf16(
                /*neg_a=*/false, fa.v, /*neg_b=*/false, fb[j].v,
                /*c_mod=*/(short)0, acc[j], /*reuse_a=*/false, /*reuse_b=*/false);
        }

#if __has_builtin(__builtin_amdgcn_sched_group_barrier)
        // Pin per-iteration pipeline: DS-write, VMEM prefetch, DS-reads, WMMAs.
        __builtin_amdgcn_sched_group_barrier(0x200, 1,  0);  // 1  DS write (A stage)
        __builtin_amdgcn_sched_group_barrier(0x020, 2,  0);  // 2  VMEM reads (A prefetch)
        __builtin_amdgcn_sched_group_barrier(0x100, 10, 0);  // 10 DS reads (fragments)
        __builtin_amdgcn_sched_group_barrier(0x008, 4,  0);  // 4  WMMA
#endif
        if (i + 1 < iters) wait_async0();   // own B-DMA for tile i+1 done
        __syncthreads();                    // everyone's tile i+1 visible
    }

    // ---- epilogue: C/D layout — VGPR r holds (M = r + 8*half, N = l16) -----
#pragma unroll
    for (int j = 0; j < 4; ++j) {
#pragma unroll
        for (int r = 0; r < 8; ++r) {
            float v = acc[j][r];
            const int m = rg * 16 + r + 8 * half;
            const int n = (cg * 4 + j) * 16 + l16;
            if (RELU) {
                Cb[(long)m * Ntot + n] = fmaxf(v, 0.0f);        // h: reused, keep cached
            } else {
                __builtin_nontemporal_store(v, &Cb[(long)m * Ntot + n]); // out: write-once
            }
        }
    }
}

// ---------------------------------------------------------------------------
extern "C" void kernel_launch(void* const* d_in, const int* in_sizes, int n_in,
                              void* d_out, int out_size, void* d_ws, size_t ws_size,
                              hipStream_t stream) {
    (void)in_sizes; (void)n_in; (void)out_size; (void)ws_size;

    const float* x  = (const float*)d_in[0];   // [8,2048,1024]
    const float* kv = (const float*)d_in[1];   // [8,128]
    const float* Dm = (const float*)d_in[2];   // [131072,128]
    const float* Um = (const float*)d_in[3];   // [131072,128]
    float* out = (float*)d_out;                // [8,2048,1024]

    // workspace carve-up (12 MB total)
    unsigned short* Dt = (unsigned short*)d_ws;                 // [8][128][1024] bf16, 2 MB
    unsigned short* Ut = Dt + (size_t)BATCH * LOW * IN_DIM;     // [8][1024][128] bf16, 2 MB
    float*          h  = (float*)(Ut + (size_t)BATCH * OUT_DIM * LOW);  // [8][2048][128] f32, 8 MB

    // 1) hypernet weight generation (memory bound: streams 128 MB of D/U)
    gen_weights_kernel<<<dim3((2 * IN_DIM * LOW) / 256), 256, 0, stream>>>(Dm, Um, kv, Dt, Ut);

    // 2) h = relu(x @ D[b]) : M=2048/batch, N=128, K=1024
    wmma_gemm_kernel<true><<<dim3(SEQ / BM, LOW / BN, BATCH), 512, 0, stream>>>(
        x, Dt, h, IN_DIM, LOW);

    // 3) out = h @ U[b] : M=2048/batch, N=1024, K=128
    wmma_gemm_kernel<false><<<dim3(SEQ / BM, OUT_DIM / BN, BATCH), 512, 0, stream>>>(
        h, Ut, out, LOW, OUT_DIM);
}
